// DeformAttn_5789615915446
// MI455X (gfx1250) — matrix operand order repaired and checked by reading, not compiled, not run on hardware
//
#include <hip/hip_runtime.h>
#include <utility>

typedef __attribute__((ext_vector_type(16))) _Float16     v16h;
typedef __attribute__((ext_vector_type(8)))  float        v8f;
typedef __attribute__((ext_vector_type(4)))  unsigned int v4u;

#define IN_C   128
#define OUT_C  128
#define NS     9
#define DPIX   16384      // FH*FW
#define BATCH  4
#define ATT_SCALE 0.25f   // HEAD_C^-0.5

// ---------- compile-time for loop (C++14-safe) ----------
template <int I, int N, class F>
struct StaticFor {
    __device__ __forceinline__ static void run(F& f) {
        f(std::integral_constant<int, I>{});
        StaticFor<I + 1, N, F>::run(f);
    }
};
template <int N, class F>
struct StaticFor<N, N, F> { __device__ __forceinline__ static void run(F&) {} };
template <int N, class F>
__device__ __forceinline__ void static_for(F&& f) { StaticFor<0, N, F>::run(f); }

// ---------- opaque LDS loads with immediate offsets ----------
// asm volatile: not hoistable/CSE-able; "n" template immediate: no address VGPRs.
template <int OFF>
__device__ __forceinline__ v16h lds_load_a(unsigned base) {
    static_assert(OFF + 16 <= 65535, "ds offset overflow");
    v4u lo, hi;
    asm volatile("ds_load_b128 %0, %2 offset:%3\n\t"
                 "ds_load_b128 %1, %2 offset:%4\n\t"
                 "s_wait_dscnt 0x0"
                 : "=v"(lo), "=v"(hi)
                 : "v"(base), "n"(OFF), "n"(OFF + 16));
    union { v4u u[2]; v16h h; } c; c.u[0] = lo; c.u[1] = hi; return c.h;
}
template <int OFF>
__device__ __forceinline__ v8f lds_load_c(unsigned base) {
    static_assert(OFF + 16 <= 65535, "ds offset overflow");
    v4u lo, hi;
    asm volatile("ds_load_b128 %0, %2 offset:%3\n\t"
                 "ds_load_b128 %1, %2 offset:%4\n\t"
                 "s_wait_dscnt 0x0"
                 : "=v"(lo), "=v"(hi)
                 : "v"(base), "n"(OFF), "n"(OFF + 16));
    union { v4u u[2]; v8f f; } c; c.u[0] = lo; c.u[1] = hi; return c.f;
}

// Dynamic LDS layout:
//   [0 .. 1536)          : biases, 3*128 f32  (bq, bk, bv)
//   [1536 .. 1536+98304) : f16 weights, pre-swizzled into WMMA A-operand
//                          layout: [mat][m][kc][lane][16 halves] (32B/lane/tile)
extern __shared__ char smem_raw[];

__global__ __launch_bounds__(256, 1)
void deform_attn_kernel(const float* __restrict__ q,  const float* __restrict__ kv,
                        const float* __restrict__ Wq, const float* __restrict__ bq,
                        const float* __restrict__ Wk, const float* __restrict__ bk,
                        const float* __restrict__ Wv, const float* __restrict__ bv,
                        float* __restrict__ out, int n_tiles)
{
    float*    sBias = (float*)smem_raw;                // 384 floats
    _Float16* sW    = (_Float16*)(smem_raw + 1536);    // 49152 halves

    const int tid = threadIdx.x;

    // ---- stage biases ----
    for (int i = tid; i < 3 * 128; i += 256) {
        const float* src = (i < 128) ? bq : (i < 256 ? bk : bv);
        sBias[i] = src[i & 127];
    }
    // ---- stage weights, f32->f16, swizzled into A-operand lane layout ----
    // A 16x32 f16 k order used here (contiguous; B uses the same): lanes 0-15 (M=lane): h[0..15]=K 0..15
    //                     lanes 16-31 (M=lane-16): h[0..15]=K 16..31
    for (int i = tid; i < 3 * 8 * 4 * 32 * 16; i += 256) {
        int e    = i & 15;
        int lane = (i >> 4) & 31;
        int kc   = (i >> 9) & 3;
        int m    = (i >> 11) & 7;
        int mat  = i >> 14;
        int half = lane >> 4;
        int ch   = kc * 32 + half * 16 + e;
        int row  = m * 16 + (lane & 15);
        const float* Wsrc = (mat == 0) ? Wq : (mat == 1 ? Wk : Wv);
        sW[i] = (_Float16)Wsrc[row * IN_C + ch];
    }
    __syncthreads();

    const int lane  = tid & 31;
    const int wave  = tid >> 5;
    const int half  = lane >> 4;
    const int px    = lane & 15;
    const int choff = half * 16;   // B-operand: channel offset inside 32-K chunk
    const int roff  = half * 8;    // C-operand: row (channel) offset inside 16-M tile

    // Only 4 live LDS base addresses; all tile selects are ds immediate offsets.
    const unsigned ldsBias = (unsigned)(size_t)sBias + (unsigned)roff * 4u;
    const unsigned ldsWq   = (unsigned)(size_t)sW + (unsigned)lane * 32u;
    const unsigned ldsWk   = ldsWq + 32768u;
    const unsigned ldsWv   = ldsWq + 65536u;

    const int waves_total = gridDim.x * 8;

    for (int t = blockIdx.x * 8 + wave; t < n_tiles; t += waves_total) {
        const int gp0 = t << 4;
        const int b   = gp0 >> 14;          // 16384 pixels per batch image
        const int d0  = gp0 & (DPIX - 1);

        // ---- q B-tiles (K=128 as 4 chunks of 32), f32 -> f16 ----
        v16h Bq4[4];
        #pragma unroll
        for (int kc = 0; kc < 4; ++kc) {
            const float* base = q + ((size_t)(b * IN_C + kc * 32 + choff)) * DPIX + d0 + px;
            union { v16h v; _Float16 h[16]; } u;
            #pragma unroll
            for (int j = 0; j < 16; ++j) u.h[j] = (_Float16)base[(size_t)j * DPIX];
            Bq4[kc] = u.v;
        }

        // ---- qp = Wq*q + bq : kc-outer / m-inner (8 independent WMMAs per chunk,
        //      each A-tile consumed immediately -> no run-ahead spills) ----
        v8f qp[8];
        static_for<8>([&](auto mc) {
            constexpr int m = decltype(mc)::value;
            qp[m] = lds_load_c<m * 64>(ldsBias);          // bq chunk as C init
        });
        static_for<4>([&](auto kcc) {
            constexpr int kc = decltype(kcc)::value;
            static_for<8>([&](auto mc) {
                constexpr int m = decltype(mc)::value;
                v16h a = lds_load_a<(m * 4 + kc) * 1024>(ldsWq);
                qp[m] = __builtin_amdgcn_wmma_f32_16x16x32_f16(false, a, false, Bq4[kc],
                                                               (short)0, qp[m], false, false);
            });
        });

        // ---- dotk[m] = qp . bk  (sample-invariant part of the logit) ----
        float dotk[8];
        static_for<8>([&](auto mc) {
            constexpr int m = decltype(mc)::value;
            v8f bkt = lds_load_c<512 + m * 64>(ldsBias);
            float tb = 0.0f;
            #pragma unroll
            for (int r = 0; r < 8; ++r) tb += qp[m][r] * bkt[r];
            tb += __shfl_xor(tb, 16, 32);
            dotk[m] = tb;
        });

        v8f   acc[8];
        float lsum[8];
        #pragma unroll
        for (int m = 0; m < 8; ++m) { acc[m] = (v8f){}; lsum[m] = 0.0f; }

        // ---- stream the 9 samples: kp/vp live only in registers ----
        for (int s = 0; s < NS; ++s) {
            v16h Bs[4];
            #pragma unroll
            for (int kc = 0; kc < 4; ++kc) {
                const float* base = kv + ((size_t)((b * IN_C + kc * 32 + choff) * NS + s)) * DPIX
                                       + d0 + px;
                if (s + 1 < NS) __builtin_prefetch(base + DPIX, 0, 1);  // next sample's lines
                union { v16h v; _Float16 h[16]; } u;
                #pragma unroll
                for (int j = 0; j < 16; ++j) u.h[j] = (_Float16)base[(size_t)j * (size_t)(NS * DPIX)];
                Bs[kc] = u.v;
            }
            // m-outer keeps only ck/cv transient (acc+qp already occupy 128 regs);
            // k/v chains interleaved per K-chunk for dependency distance 2.
            static_for<8>([&](auto mc) {
                constexpr int m = decltype(mc)::value;
                v8f ck = (v8f){};   // k bias handled via dotk[m]
                v8f cv = (v8f){};   // v bias handled in epilogue (softmax sums to 1)
                static_for<4>([&](auto kcc) {
                    constexpr int kc = decltype(kcc)::value;
                    v16h ak = lds_load_a<(m * 4 + kc) * 1024>(ldsWk);
                    ck = __builtin_amdgcn_wmma_f32_16x16x32_f16(false, ak, false, Bs[kc],
                                                                (short)0, ck, false, false);
                    v16h av = lds_load_a<(m * 4 + kc) * 1024>(ldsWv);
                    cv = __builtin_amdgcn_wmma_f32_16x16x32_f16(false, av, false, Bs[kc],
                                                                (short)0, cv, false, false);
                });
                // logit: per-lane 8-channel partial dot + cross-half add (same pixel)
                float tp = 0.0f;
                #pragma unroll
                for (int r = 0; r < 8; ++r) tp += qp[m][r] * ck[r];
                tp += __shfl_xor(tp, 16, 32);
                float p = __expf((tp + dotk[m]) * ATT_SCALE);
                lsum[m] += p;
                #pragma unroll
                for (int r = 0; r < 8; ++r) acc[m][r] += p * cv[r];
            });
        }

        // ---- normalize, add bv (softmax weights sum to 1), store ----
        static_for<8>([&](auto mc) {
            constexpr int m = decltype(mc)::value;
            float inv = 1.0f / lsum[m];
            v8f bvt = lds_load_c<1024 + m * 64>(ldsBias);
            float* obase = out + ((size_t)(b * OUT_C + m * 16 + roff)) * DPIX + d0 + px;
            #pragma unroll
            for (int r = 0; r < 8; ++r) obase[(size_t)r * DPIX] = acc[m][r] * inv + bvt[r];
        });
    }
}

extern "C" void kernel_launch(void* const* d_in, const int* in_sizes, int n_in,
                              void* d_out, int out_size, void* d_ws, size_t ws_size,
                              hipStream_t stream) {
    (void)in_sizes; (void)n_in; (void)out_size; (void)d_ws; (void)ws_size;
    const float* q  = (const float*)d_in[0];
    const float* kv = (const float*)d_in[1];
    const float* Wq = (const float*)d_in[2];
    const float* bq = (const float*)d_in[3];
    const float* Wk = (const float*)d_in[4];
    const float* bk = (const float*)d_in[5];
    const float* Wv = (const float*)d_in[6];
    const float* bv = (const float*)d_in[7];
    float* out = (float*)d_out;

    const int n_tiles = (BATCH * DPIX) / 16;               // 4096 (16-pixel tiles)
    const size_t shmem = 1536 + (size_t)3 * 8 * 4 * 32 * 16 * 2;  // biases + f16 weights
    // 256 blocks * 8 waves = 2048 waves -> exactly 2 tiles per wave (no divergence)
    hipLaunchKernelGGL(deform_attn_kernel, dim3(256), dim3(256), shmem, stream,
                       q, kv, Wq, bq, Wk, bk, Wv, bv, out, n_tiles);
}
